// SheafBuilderGeneral_67980742361299
// MI455X (gfx1250) — matrix operand order repaired and checked by reading, not compiled, not run on hardware
//
#include <hip/hip_runtime.h>
#include <hip/hip_bf16.h>
#include <stdint.h>

typedef __attribute__((ext_vector_type(16))) __bf16 v16bf;
typedef __attribute__((ext_vector_type(8)))  __bf16 v8bf;
typedef __attribute__((ext_vector_type(4)))  __bf16 v4bf;
typedef __attribute__((ext_vector_type(8)))  float  v8f;

#define F 128          // feature dim (reference f)
#define TWO_F 256      // 2*f = K of the GEMM
#define DH 4           // heads D
#define DD 16          // D*D = N of the GEMM

// ---------------------------------------------------------------------------
// Kernel 1: mean over the D=4 replicated rows, convert f32 -> bf16.
// ---------------------------------------------------------------------------
__global__ __launch_bounds__(256) void mean_to_bf16_kernel(
    const float* __restrict__ in, __bf16* __restrict__ out, int nRows) {
  int gid = blockIdx.x * blockDim.x + threadIdx.x;
  int total = nRows * (F / 4);
  if (gid >= total) return;
  int n  = gid >> 5;          // row (F/4 == 32 quads per row)
  int kq = (gid & 31) << 2;   // feature offset (multiple of 4)
  const float4* p = (const float4*)(in + (size_t)n * (4 * F) + kq);
  float4 a = p[0 * (F / 4)];
  float4 b = p[1 * (F / 4)];
  float4 c = p[2 * (F / 4)];
  float4 d = p[3 * (F / 4)];
  v4bf o;
  o[0] = (__bf16)((a.x + b.x + c.x + d.x) * 0.25f);
  o[1] = (__bf16)((a.y + b.y + c.y + d.y) * 0.25f);
  o[2] = (__bf16)((a.z + b.z + c.z + d.z) * 0.25f);
  o[3] = (__bf16)((a.w + b.w + c.w + d.w) * 0.25f);
  *(v4bf*)(out + (size_t)n * F + kq) = o;
}

// ---------------------------------------------------------------------------
// Kernel 2: convert W [16, 256] f32 -> bf16 (row-major, becomes WMMA B panels).
// ---------------------------------------------------------------------------
__global__ __launch_bounds__(256) void w_to_bf16_kernel(
    const float* __restrict__ w, __bf16* __restrict__ wb, int n) {
  int gid = blockIdx.x * blockDim.x + threadIdx.x;
  if (gid < n) wb[gid] = (__bf16)w[gid];
}

// Build a 16x32 bf16 A fragment for this lane from a contiguous bf16 row.
// Lanes 0-15 hold K {0..15}; lanes 16-31 hold K {16..31}.
static __device__ inline v16bf load_a_frag(const __bf16* chunk_base, int hi) {
  v16bf a = *(const v16bf*)(chunk_base + 16 * hi);
  return a;
}

static __device__ inline float fast_sigmoid(float t) {
  return __builtin_amdgcn_rcpf(1.0f + __expf(-t));   // v_exp_f32 + v_rcp_f32
}

// ---------------------------------------------------------------------------
// Kernel 3: main. One wave = one tile of 16 nonzeros:
//   gather 16x256 bf16 (xm||em), 8x v_wmma_f32_16x16x32_bf16 against
//   loop-invariant W panels, sigmoid, scatter attrs + int64 indices.
// All indexing is 32-bit (nnz*16 = 16M << 2^31); int64 inputs read low-word.
// ---------------------------------------------------------------------------
__global__ __launch_bounds__(256) void sheaf_wmma_kernel(
    const __bf16* __restrict__ xm, const __bf16* __restrict__ em,
    const __bf16* __restrict__ wb, const float* __restrict__ bias,
    const int* __restrict__ rowp2,  // int64 row indices viewed as int pairs
    const int* __restrict__ colp2,  // int64 col indices viewed as int pairs
    long long* __restrict__ out_idx, float* __restrict__ out_attr,
    int nnz, int tiles) {
  const int lane = threadIdx.x & 31;
  const int hi   = lane >> 4;    // half-wave select
  const int n15  = lane & 15;    // N column (and entry slot for gathers)
  const int waveId = blockIdx.x * (blockDim.x >> 5) + (threadIdx.x >> 5);
  const int nWaves = gridDim.x * (blockDim.x >> 5);
  const unsigned half = (unsigned)nnz * DD;   // second index row offset

  // Loop-invariant B operand: B[k][n] = W[n][k]; N = lane&15, contiguous K
  // range per half-wave -> one 32-byte load per K-chunk.
  v16bf Breg[8];
#pragma unroll
  for (int c = 0; c < 8; ++c)
    Breg[c] = *(const v16bf*)(wb + (size_t)n15 * TWO_F + 32 * c + 16 * hi);
  const float bv = bias[n15];

  for (int t = waveId; t < tiles; t += nWaves) {
    const int tbase = t * 16;
    int eIdx = tbase + n15;
    if (eIdx >= nnz) eIdx = nnz - 1;            // clamp tail (writes predicated)
    const int rlow = rowp2[2 * eIdx];           // low word of int64 (value < 2^31)
    const int clow = colp2[2 * eIdx];
    const __bf16* xrow = xm + (unsigned)rlow * F;
    const __bf16* erow = em + (unsigned)clow * F;

    v8f acc;                                    // fuse bias into C init
#pragma unroll
    for (int r = 0; r < 8; ++r) acc[r] = bv;

#pragma unroll
    for (int c = 0; c < 4; ++c) {               // K = 0..127 from xm
      v16bf a = load_a_frag(xrow + 32 * c, hi);
      acc = __builtin_amdgcn_wmma_f32_16x16x32_bf16(
          false, a, false, Breg[c], (short)0, acc, false, false);
    }
#pragma unroll
    for (int c = 0; c < 4; ++c) {               // K = 128..255 from em
      v16bf a = load_a_frag(erow + 32 * c, hi);
      acc = __builtin_amdgcn_wmma_f32_16x16x32_bf16(
          false, a, false, Breg[4 + c], (short)0, acc, false, false);
    }

    if (tbase + 16 <= nnz) {                    // fast path: full tile, no guards
#pragma unroll
      for (int r = 0; r < 8; ++r) {
        const int M  = r + 8 * hi;              // entry slot for this acc register
        const int rv = __shfl(rlow, M, 32);
        const int cv = __shfl(clow, M, 32);
        const unsigned j = (unsigned)(tbase + M) * DD + n15;
        out_attr[j]       = fast_sigmoid(acc[r]);
        out_idx[j]        = (long long)(4 * rv + (n15 >> 2));
        out_idx[half + j] = (long long)(4 * cv + (n15 & 3));
      }
    } else {                                    // tail tile: per-entry guard
#pragma unroll
      for (int r = 0; r < 8; ++r) {
        const int M  = r + 8 * hi;
        const int rv = __shfl(rlow, M, 32);     // full-EXEC shuffle first
        const int cv = __shfl(clow, M, 32);
        const int entry = tbase + M;
        if (entry < nnz) {
          const unsigned j = (unsigned)entry * DD + n15;
          out_attr[j]       = fast_sigmoid(acc[r]);
          out_idx[j]        = (long long)(4 * rv + (n15 >> 2));
          out_idx[half + j] = (long long)(4 * cv + (n15 & 3));
        }
      }
    }
  }
}

// ---------------------------------------------------------------------------
extern "C" void kernel_launch(void* const* d_in, const int* in_sizes, int n_in,
                              void* d_out, int out_size, void* d_ws, size_t ws_size,
                              hipStream_t stream) {
  const float* x    = (const float*)d_in[0];
  const float* e    = (const float*)d_in[1];
  const int*   hidx = (const int*)d_in[2];     // int64 pairs viewed as ints
  const float* W    = (const float*)d_in[3];
  const float* bias = (const float*)d_in[4];

  const long long N   = (long long)in_sizes[0] / (DH * F);  // 100000
  const long long E   = (long long)in_sizes[1] / (DH * F);  // 20000
  const long long nnz = (long long)in_sizes[2] / 2;         // 1000000
  const int wElems    = in_sizes[3];                        // 16*256

  // Workspace: xm (N*128 bf16) | em (E*128 bf16) | W bf16 (4096)
  __bf16* xm = (__bf16*)d_ws;
  __bf16* em = xm + (size_t)N * F;
  __bf16* wb = em + (size_t)E * F;

  // Output: h_index int64 [2, nnz*16] then h_attrs float [nnz*16]
  long long* out_idx  = (long long*)d_out;
  float*     out_attr = (float*)(out_idx + 2LL * nnz * DD);

  {
    int total = (int)(N * (F / 4));
    mean_to_bf16_kernel<<<(total + 255) / 256, 256, 0, stream>>>(x, xm, (int)N);
  }
  {
    int total = (int)(E * (F / 4));
    mean_to_bf16_kernel<<<(total + 255) / 256, 256, 0, stream>>>(e, em, (int)E);
  }
  w_to_bf16_kernel<<<(wElems + 255) / 256, 256, 0, stream>>>(W, wb, wElems);

  const int tiles = (int)((nnz + 15) / 16);                 // 62500
  int blocks = (tiles + 7) / 8;                             // 8 waves per block
  if (blocks > 4096) blocks = 4096;
  const int* rowp2 = hidx;                                  // int64 row words
  const int* colp2 = hidx + 2 * nnz;                        // int64 col words
  sheaf_wmma_kernel<<<blocks, 256, 0, stream>>>(
      xm, em, wb, bias, rowp2, colp2, out_idx, out_attr, (int)nnz, tiles);
}
